// RestrictedSelfAttention_41403484733606
// MI455X (gfx1250) — matrix operand order repaired and checked by reading, not compiled, not run on hardware
//
#include <hip/hip_runtime.h>
#include <hip/hip_bf16.h>
#include <cstdint>

// ---------------------------------------------------------------------------
// RestrictedSelfAttention for MI455X (gfx1250), wave32 + WMMA bf16 +
// async global->LDS staging (ASYNCcnt path).
//   out  = (ctx @ Wo^T + bo)            [B,S,E]  f32  -> d_out[0 .. 3145728)
//   attn = softmax(banded scores)+1e-6  [B,H,S,S] f32 -> d_out[3145728 .. )
// ---------------------------------------------------------------------------

typedef __bf16 bf16_t;
typedef __attribute__((ext_vector_type(16))) __bf16 v16bf;
typedef __attribute__((ext_vector_type(8)))  __bf16 v8bf;
typedef __attribute__((ext_vector_type(8)))  float  v8f;

#define EMB  768
#define NB   2
#define SEQ  2048
#define NH   12
#define HD   64
#define NTOK (NB * SEQ)                          // 4096
#define WIN  64
#define WELEM (EMB * EMB)                        // 589824
#define QKV_ELEM ((size_t)NTOK * EMB)            // 3145728
#define ATTN_ELEM ((size_t)NB * NH * SEQ * SEQ)  // 100663296
#define ALD 776                                  // padded LDS A-tile row stride (elems)

__device__ __forceinline__ v8f wmma_bf16(v16bf a, v16bf b, v8f c) {
  return __builtin_amdgcn_wmma_f32_16x16x32_bf16(false, a, false, b, (short)0, c,
                                                 false, false);
}

// A-fragment (16x32, M x K), layout used here: lane L holds row M=L%16;
// elems 0..15 -> K = (L/16)*16 + e (contiguous, matches B fragment).
__device__ __forceinline__ v16bf load_a_frag(const bf16_t* __restrict__ p, int ld,
                                             int m0, int k0) {
  int lane = threadIdx.x & 31;
  const bf16_t* r = p + (size_t)(m0 + (lane & 15)) * ld + k0 + ((lane >> 4) << 4);
  v8bf lo = *(const v8bf*)r;
  v8bf hi = *(const v8bf*)(r + 8);
  v16bf a;
#pragma unroll
  for (int i = 0; i < 8; ++i) { a[i] = lo[i]; a[i + 8] = hi[i]; }
  return a;
}

// B-fragment (32x16, K x N) where B = src^T and src is row-major [n, k], stride ld.
// layout used here: lane L holds col N=L%16, K = k0 + (L/16)*16 + e (e=0..15 contiguous).
__device__ __forceinline__ v16bf load_bT_frag(const bf16_t* __restrict__ p, int ld,
                                              int n0, int k0) {
  int lane = threadIdx.x & 31;
  return *(const v16bf*)(p + (size_t)(n0 + (lane & 15)) * ld + k0 +
                         ((lane >> 4) << 4));
}

// ---------------------------------------------------------------------------
__global__ void f32_to_bf16_kernel(const float* __restrict__ src,
                                   bf16_t* __restrict__ dst, int n) {
  int i = blockIdx.x * blockDim.x + threadIdx.x;
  if (i < n) dst[i] = (bf16_t)src[i];
}

// Y = X @ W^T + bias, X bf16 [NTOK, EMB]. Block = 8 waves: one 16-token row
// tile (staged to LDS via async copy) x 8 consecutive 16-col tiles.
// mode 0: Q -> [B,H,S,D] bf16   mode 1: K -> [B,H,S,D] bf16
// mode 2: V -> [B,H,D,S] bf16   mode 3: out -> f32 with NaN guard
__global__ void gemm_proj_kernel(const bf16_t* __restrict__ X,
                                 const bf16_t* __restrict__ Wbf,
                                 const float* __restrict__ bias, int mode,
                                 bf16_t* __restrict__ dst_bf,
                                 float* __restrict__ dst_f32,
                                 const float* __restrict__ vguard) {
  __shared__ bf16_t Atile[16 * ALD];  // 16 x 768 bf16, padded rows (~24.8 KB)
  int tid = threadIdx.x;
  int warp = tid >> 5, lane = tid & 31;
  int mt = blockIdx.x;                 // 0..255 (token tiles)
  int m0 = mt * 16;
  int nt = blockIdx.y * 8 + warp;      // 0..47  (output-col tiles)

  // ---- async-stage A tile: 16 rows x 768 bf16 = 1536 x 16B chunks --------
  uint32_t ldsBase = (uint32_t)(uintptr_t)(void*)&Atile[0];  // low 32b = LDS offset
#pragma unroll
  for (int i = 0; i < 6; ++i) {
    int c = tid + i * 256;             // chunk id 0..1535
    int row = c / 96;                  // 96 x 16B chunks per row
    int cc = c % 96;
    uint64_t gaddr = (uint64_t)(uintptr_t)(X + (size_t)(m0 + row) * EMB + cc * 8);
    uint32_t laddr = ldsBase + (uint32_t)((row * ALD + cc * 8) * 2);
    asm volatile("global_load_async_to_lds_b128 %0, %1, off"
                 :: "v"(laddr), "v"(gaddr) : "memory");
  }
  asm volatile("s_wait_asynccnt 0x0" ::: "memory");
  __syncthreads();

  // ---- K-loop: 24 x v_wmma_f32_16x16x32_bf16 -----------------------------
  v8f acc = {0.f, 0.f, 0.f, 0.f, 0.f, 0.f, 0.f, 0.f};
#pragma unroll 4
  for (int k0 = 0; k0 < EMB; k0 += 32) {
    v16bf a = load_a_frag((const bf16_t*)Atile, ALD, 0, k0);
    v16bf b = load_bT_frag(Wbf, EMB, nt * 16, k0);
    if (k0 + 32 < EMB)  // global_prefetch_b8 of next W chunk
      __builtin_prefetch((const void*)(Wbf + (size_t)(nt * 16 + (lane & 15)) * EMB + k0 + 32), 0, 0);
    acc = wmma_bf16(a, b, acc);
  }

  int n = nt * 16 + (lane & 15);
  int h = n >> 6, d = n & 63;
  float bv = bias[n];
#pragma unroll
  for (int j = 0; j < 8; ++j) {
    int r = m0 + j + ((lane >> 4) << 3);  // token index (C layout row)
    float val = acc[j] + bv;
    if (mode <= 1) {
      int b_ = r >> 11, s = r & (SEQ - 1);
      dst_bf[(((size_t)(b_ * NH + h) * SEQ) + s) * HD + d] = (bf16_t)val;
    } else if (mode == 2) {
      int b_ = r >> 11, s = r & (SEQ - 1);
      dst_bf[(((size_t)(b_ * NH + h) * HD) + d) * SEQ + s] = (bf16_t)val;
    } else {
      if (val != val) val = vguard[(size_t)r * EMB + n];  // NaN guard
      dst_f32[(size_t)r * EMB + n] = val;
    }
  }
}

// colsumV[b,h,d] = sum_s V[b,h,d,s]   (for the +1e-6 * sum_k v_k term)
__global__ void colsum_v_kernel(const bf16_t* __restrict__ Vt,
                                float* __restrict__ colsum) {
  int i = blockIdx.x * blockDim.x + threadIdx.x;  // over B*H*D = 1536
  if (i >= NB * NH * HD) return;
  const bf16_t* row = Vt + (size_t)i * SEQ;
  float s = 0.f;
  for (int k = 0; k < SEQ; k += 8) {
    v8bf v = *(const v8bf*)(row + k);
#pragma unroll
    for (int e = 0; e < 8; ++e) s += (float)v[e];
  }
  colsum[i] = s;
}

__global__ void fill_attn_kernel(float4* __restrict__ dst, long long n4) {
  long long i = (long long)blockIdx.x * blockDim.x + threadIdx.x;
  if (i < n4) dst[i] = make_float4(1e-6f, 1e-6f, 1e-6f, 1e-6f);
}

// One wave per (b, h, 16-query block). Band of 9 key tiles (144 keys).
__global__ void attn_kernel(const bf16_t* __restrict__ Q,
                            const bf16_t* __restrict__ K,
                            const bf16_t* __restrict__ Vt,
                            const unsigned char* __restrict__ pad_mask,
                            const float* __restrict__ colsum,
                            float* __restrict__ attn_out,
                            bf16_t* __restrict__ ctx) {
  __shared__ bf16_t pbuf[8][16][160];  // per-wave 16x160 bf16 P tile (40 KB)
  int warp = threadIdx.x >> 5, lane = threadIdx.x & 31;
  int w = blockIdx.x * 8 + warp;       // 0 .. 3071
  int qb = w & 127;
  int h = (w >> 7) % NH;
  int b = w / (128 * NH);
  int q0 = qb * 16;
  const bf16_t* Qh = Q + ((size_t)(b * NH + h) * SEQ) * HD;
  const bf16_t* Kh = K + ((size_t)(b * NH + h) * SEQ) * HD;
  const bf16_t* Vh = Vt + ((size_t)(b * NH + h) * HD) * SEQ;

  // ---- scores: S(16x144) = Q(16x64) x K^T, 2 WMMAs (K=32) per key tile ----
  v16bf qa0 = load_a_frag(Qh, HD, q0, 0);
  v16bf qa1 = load_a_frag(Qh, HD, q0, 32);
  v8f acc[9];
#pragma unroll
  for (int t = 0; t < 9; ++t) {
    int kti = qb - 4 + t;
    v8f c = {0.f, 0.f, 0.f, 0.f, 0.f, 0.f, 0.f, 0.f};
    if (kti >= 0 && kti < SEQ / 16) {
      v16bf kb0 = load_bT_frag(Kh, HD, kti * 16, 0);
      v16bf kb1 = load_bT_frag(Kh, HD, kti * 16, 32);
      c = wmma_bf16(qa0, kb0, c);
      c = wmma_bf16(qa1, kb1, c);
    }
    acc[t] = c;
  }

  // ---- scale + masks ----
  const float scale = 0.125f;  // 1/sqrt(64)
#pragma unroll
  for (int t = 0; t < 9; ++t) {
    int kti = qb - 4 + t;
    int k = kti * 16 + (lane & 15);
    bool kin = (kti >= 0 && kti < SEQ / 16);
    bool padded = kin ? (pad_mask[b * SEQ + k] != 0) : false;
#pragma unroll
    for (int j = 0; j < 8; ++j) {
      int q = q0 + j + ((lane >> 4) << 3);
      float v = acc[t][j] * scale;
      if (v != v) v = -10000.0f;  // NaN -> -10000 (reference semantics)
      int dqk = q - k; if (dqk < 0) dqk = -dqk;
      if (!kin || padded || dqk > WIN) v = -10000.0f;
      acc[t][j] = v;
    }
  }

  // ---- softmax per row: reduce across 16 lanes of each half-wave ----
#pragma unroll
  for (int j = 0; j < 8; ++j) {
    float m = -3.0e38f;
#pragma unroll
    for (int t = 0; t < 9; ++t) m = fmaxf(m, acc[t][j]);
#pragma unroll
    for (int off = 1; off < 16; off <<= 1) m = fmaxf(m, __shfl_xor(m, off, 32));
    float s = 0.f;
#pragma unroll
    for (int t = 0; t < 9; ++t) {
      float p = __expf(acc[t][j] - m);  // masked: exp(~-10000) underflows to 0
      acc[t][j] = p;
      s += p;
    }
#pragma unroll
    for (int off = 1; off < 16; off <<= 1) s += __shfl_xor(s, off, 32);
    float inv = 1.0f / s;  // s >= 1 (diagonal always unmasked)
#pragma unroll
    for (int t = 0; t < 9; ++t) acc[t][j] *= inv;
  }

  // ---- write band of attn (+1e-6) and stage P into LDS (bf16) ----
  bf16_t(*pb)[160] = pbuf[warp];
  float* attnBase = attn_out + (size_t)(b * NH + h) * SEQ * SEQ;
#pragma unroll
  for (int t = 0; t < 9; ++t) {
    int kti = qb - 4 + t;
    int col = t * 16 + (lane & 15);
    int k = kti * 16 + (lane & 15);
    bool kin = (kti >= 0 && kti < SEQ / 16);
#pragma unroll
    for (int j = 0; j < 8; ++j) {
      int row = j + ((lane >> 4) << 3);
      float p = acc[t][j];
      pb[row][col] = (bf16_t)p;
      if (kin) attnBase[(size_t)(q0 + row) * SEQ + k] = p + 1e-6f;
    }
  }
  {  // zero the K-pad columns 144..159 so the 5th WMMA chunk is benign
    int col = 144 + (lane & 15);
#pragma unroll
    for (int j = 0; j < 8; ++j) pb[j + ((lane >> 4) << 3)][col] = (bf16_t)0.f;
  }
  __syncthreads();

  // ---- ctx(16x64) = P(16x160) x V(160x64), 5 K-chunks x 4 N-tiles ----
  v16bf pa[5];
#pragma unroll
  for (int kk = 0; kk < 5; ++kk)
    pa[kk] = load_a_frag(&pb[0][0], 160, 0, kk * 32);

#pragma unroll
  for (int nt = 0; nt < 4; ++nt) {
    v8f c = {0.f, 0.f, 0.f, 0.f, 0.f, 0.f, 0.f, 0.f};
#pragma unroll
    for (int kk = 0; kk < 5; ++kk) {
      int kbase = (qb - 4) * 16 + kk * 32 + ((lane >> 4) << 4);
      kbase = kbase < 0 ? 0 : (kbase > SEQ - 16 ? SEQ - 16 : kbase);  // P==0 there
      v16bf vb = *(const v16bf*)(Vh + (size_t)(nt * 16 + (lane & 15)) * SEQ + kbase);
      c = wmma_bf16(pa[kk], vb, c);
    }
    int d = nt * 16 + (lane & 15);
    float cs = 1e-6f * colsum[(b * NH + h) * HD + d];  // +1e-6 * sum_k v_k
#pragma unroll
    for (int j = 0; j < 8; ++j) {
      int q = q0 + j + ((lane >> 4) << 3);
      ctx[((size_t)(b * SEQ + q)) * EMB + h * HD + d] = (bf16_t)(c[j] + cs);
    }
  }
}

// ---------------------------------------------------------------------------
extern "C" void kernel_launch(void* const* d_in, const int* in_sizes, int n_in,
                              void* d_out, int out_size, void* d_ws, size_t ws_size,
                              hipStream_t stream) {
  const float* query = (const float*)d_in[0];
  const float* key   = (const float*)d_in[1];
  const float* value = (const float*)d_in[2];
  const unsigned char* pad = (const unsigned char*)d_in[3];
  const float* Wq = (const float*)d_in[4];
  const float* bq = (const float*)d_in[5];
  const float* Wk = (const float*)d_in[6];
  const float* bk = (const float*)d_in[7];
  const float* Wv = (const float*)d_in[8];
  const float* bv = (const float*)d_in[9];
  const float* Wo = (const float*)d_in[10];
  const float* bo = (const float*)d_in[11];
  float* out = (float*)d_out;                       // [0, NTOK*EMB) = out tensor
  float* attn = out + QKV_ELEM;                     // then [B,H,S,S] attn

  bf16_t* ws = (bf16_t*)d_ws;
  bf16_t* Wq_bf = ws;
  bf16_t* Wk_bf = Wq_bf + WELEM;
  bf16_t* Wv_bf = Wk_bf + WELEM;
  bf16_t* Wo_bf = Wv_bf + WELEM;
  bf16_t* Xq_bf = Wo_bf + WELEM;      // bf16 copies of the activations
  bf16_t* Xk_bf = Xq_bf + QKV_ELEM;
  bf16_t* Xv_bf = Xk_bf + QKV_ELEM;
  bf16_t* Qbf   = Xv_bf + QKV_ELEM;
  bf16_t* Kbf   = Qbf + QKV_ELEM;
  bf16_t* Vtbf  = Kbf + QKV_ELEM;
  bf16_t* ctxbf = Vtbf + QKV_ELEM;
  float*  colsum = (float*)(ctxbf + QKV_ELEM);      // even bf16 offset -> 4B aligned

  // 1) f32 -> bf16 for weights and activations
  int nw = WELEM;
  f32_to_bf16_kernel<<<(nw + 255) / 256, 256, 0, stream>>>(Wq, Wq_bf, nw);
  f32_to_bf16_kernel<<<(nw + 255) / 256, 256, 0, stream>>>(Wk, Wk_bf, nw);
  f32_to_bf16_kernel<<<(nw + 255) / 256, 256, 0, stream>>>(Wv, Wv_bf, nw);
  f32_to_bf16_kernel<<<(nw + 255) / 256, 256, 0, stream>>>(Wo, Wo_bf, nw);
  int nx = (int)QKV_ELEM;
  f32_to_bf16_kernel<<<(nx + 255) / 256, 256, 0, stream>>>(query, Xq_bf, nx);
  f32_to_bf16_kernel<<<(nx + 255) / 256, 256, 0, stream>>>(key,   Xk_bf, nx);
  f32_to_bf16_kernel<<<(nx + 255) / 256, 256, 0, stream>>>(value, Xv_bf, nx);

  // 2) projections (WMMA + async LDS staging)
  dim3 gGemm(256, 6), bGemm(256);
  gemm_proj_kernel<<<gGemm, bGemm, 0, stream>>>(Xq_bf, Wq_bf, bq, 0, Qbf, nullptr, nullptr);
  gemm_proj_kernel<<<gGemm, bGemm, 0, stream>>>(Xk_bf, Wk_bf, bk, 1, Kbf, nullptr, nullptr);
  gemm_proj_kernel<<<gGemm, bGemm, 0, stream>>>(Xv_bf, Wv_bf, bv, 2, Vtbf, nullptr, nullptr);

  // 3) V column sums
  colsum_v_kernel<<<6, 256, 0, stream>>>(Vtbf, colsum);

  // 4) bulk-fill attn with 1e-6 (out-of-window value after softmax+1e-6)
  long long n4 = (long long)(ATTN_ELEM / 4);
  fill_attn_kernel<<<(unsigned)((n4 + 255) / 256), 256, 0, stream>>>((float4*)attn, n4);

  // 5) banded attention (WMMA scores + softmax + WMMA P.V): 3072 waves
  attn_kernel<<<384, 256, 0, stream>>>(Qbf, Kbf, Vtbf, pad, colsum, attn, ctxbf);

  // 6) output projection (WMMA) with NaN guard
  gemm_proj_kernel<<<gGemm, bGemm, 0, stream>>>(ctxbf, Wo_bf, bo, 3, nullptr, out, value);
}
